// SiameseSNN_38465727103169
// MI455X (gfx1250) — compile-verified, matrix operand order checked by reading, not run on hardware
//
#include <hip/hip_runtime.h>

// ---------------------------------------------------------------------------
// Siamese SNN forward for MI455X (gfx1250, wave32, WMMA).
// Dominant op: per-step FC [256 x 15776] x [15776 x 250] done with
// v_wmma_f32_16x16x32_f16 (spikes are exactly {0,1} in f16; weights f16,
// accum f32). Working set (~112 MB) fits in the 192 MB L2, so the per-step
// weight/activation re-reads never touch HBM after warm-up.
// ---------------------------------------------------------------------------

typedef _Float16 f16;
typedef _Float16 v8h  __attribute__((ext_vector_type(8)));
typedef _Float16 v16h __attribute__((ext_vector_type(16)));
typedef float    v8f  __attribute__((ext_vector_type(8)));

#define BETA   0.9f
#define THR    1.5f
#define BN_EPS 1e-5f

constexpr int  BT    = 256;        // x1 batch (128) || x2 batch (128)
constexpr int  L1    = 1981;       // conv1 output length
constexpr int  LP    = 495;        // pooled length
constexpr int  L2C   = 493;        // conv2 output length
constexpr int  KDIM  = 32 * 493;   // 15776, FC reduction dim (=493 k-steps of 32)
constexpr int  KS    = 493;
constexpr int  NSTEP = 50;

// ---------------------------------------------------------------------------
// K0: pack fc1w (f32 [250,15776]) into the per-lane WMMA B-fragment order.
// Layout: bp[nt][ks][lane][h] (16 halfs per lane = 32B coalesced load).
// k-offset for element h of lane: (h<8 ? h : h+8) + (lane<16 ? 0 : 8).
// n for lane = nt*16 + (lane&15); zero pad n in [250,256).
// ---------------------------------------------------------------------------
__global__ void k_pack_b(const float* __restrict__ fc1w, f16* __restrict__ bp) {
  long idx = (long)blockIdx.x * 256 + threadIdx.x;
  const long total = 16l * KS * 512;
  if (idx >= total) return;
  int h    = (int)(idx & 15);
  int lane = (int)((idx >> 4) & 31);
  int ks   = (int)((idx >> 9) % KS);
  int nt   = (int)(idx / ((long)KS * 512));
  int n    = nt * 16 + (lane & 15);
  int koff = (h < 8 ? h : h + 8) + (lane < 16 ? 0 : 8);
  int k    = ks * 32 + koff;
  float v  = (n < 250) ? fc1w[(long)n * KDIM + k] : 0.0f;
  bp[idx] = (f16)v;
}

// ---------------------------------------------------------------------------
// K1: conv1 (stride 4, 80 taps) + eval BN1 -> a1 [256,16,1981]  (step-invariant)
// ---------------------------------------------------------------------------
__global__ void k_conv1(const float* __restrict__ x1, const float* __restrict__ x2,
                        const float* __restrict__ c1w, const float* __restrict__ c1b,
                        const float* __restrict__ g, const float* __restrict__ sh,
                        const float* __restrict__ mn, const float* __restrict__ vr,
                        float* __restrict__ a1) {
  long idx = (long)blockIdx.x * 256 + threadIdx.x;
  const long total = (long)BT * 16 * L1;
  if (idx >= total) return;
  int o = (int)(idx % L1);
  int c = (int)((idx / L1) % 16);
  int m = (int)(idx / ((long)16 * L1));
  const float* x = (m < 128) ? (x1 + (long)m * 8000) : (x2 + (long)(m - 128) * 8000);
  const float* w = c1w + c * 80;
  float acc = 0.0f;
  int base = o * 4;
  #pragma unroll 8
  for (int k = 0; k < 80; ++k) acc += x[base + k] * w[k];
  acc += c1b[c];
  float scale = g[c] / sqrtf(vr[c] + BN_EPS);
  a1[idx] = (acc - mn[c]) * scale + sh[c];
}

// ---------------------------------------------------------------------------
// K2: zero-fill (membrane state must be zeroed every call; ws is poisoned).
// ---------------------------------------------------------------------------
__global__ void k_zero(float* __restrict__ p, long n) {
  long i = (long)blockIdx.x * 256 + threadIdx.x;
  long stride = (long)gridDim.x * 256;
  for (; i < n; i += stride) p[i] = 0.0f;
}

// ---------------------------------------------------------------------------
// K3 (fused): per batch element m:
//   phase 1: LIF1 (in-place mem1 update) + maxpool4 -> 16x495 tile in LDS
//            (element 1980 of each row is dropped by VALID pooling and
//             feeds nothing downstream, so it is skipped)
//   phase 2: conv2 (16ch, 3 taps) + BN2 + LIF2 -> spk2 (f16, exactly {0,1})
// spk2 flat index == m*KDIM + (co*493+o), matching the reference reshape.
// LDS tile: 16*495*4 = 31,680 B.
// ---------------------------------------------------------------------------
__global__ void __launch_bounds__(256)
k_step_front(const float* __restrict__ a1, float* __restrict__ mem1,
             const float* __restrict__ c2w, const float* __restrict__ c2b,
             const float* __restrict__ g, const float* __restrict__ sh,
             const float* __restrict__ mn, const float* __restrict__ vr,
             float* __restrict__ mem2, f16* __restrict__ spk2) {
  __shared__ float pool[16 * LP];
  int m = blockIdx.x;

  // ---- phase 1: LIF1 + maxpool4 into LDS ----
  for (int i = threadIdx.x; i < 16 * LP; i += 256) {
    int c = i / LP;
    int j = i - c * LP;
    long base = ((long)m * 16 + c) * L1 + (long)j * 4;
    float pmax = 0.0f;
    #pragma unroll
    for (int t = 0; t < 4; ++t) {
      float mmv = mem1[base + t];
      float r   = (mmv > THR) ? 1.0f : 0.0f;
      mmv = BETA * mmv + a1[base + t] - r * THR;
      mem1[base + t] = mmv;
      pmax = fmaxf(pmax, (mmv > THR) ? 1.0f : 0.0f);
    }
    pool[i] = pmax;
  }
  __syncthreads();

  // ---- phase 2: conv2 + BN2 + LIF2 out of LDS ----
  for (int i = threadIdx.x; i < 32 * L2C; i += 256) {
    int co = i / L2C;
    int o  = i - co * L2C;
    const float* w = c2w + co * 48;
    float acc = c2b[co];
    #pragma unroll
    for (int ci = 0; ci < 16; ++ci) {
      const float* pr = pool + ci * LP + o;
      acc += pr[0] * w[ci * 3 + 0] + pr[1] * w[ci * 3 + 1] + pr[2] * w[ci * 3 + 2];
    }
    float scale = g[co] / sqrtf(vr[co] + BN_EPS);
    float cur = (acc - mn[co]) * scale + sh[co];
    long idx = (long)m * KDIM + i;     // i == co*493 + o
    float mmv = mem2[idx];
    float r   = (mmv > THR) ? 1.0f : 0.0f;
    mmv = BETA * mmv + cur - r * THR;
    mem2[idx] = mmv;
    spk2[idx] = (f16)((mmv > THR) ? 1.0f : 0.0f);
  }
}

// ---------------------------------------------------------------------------
// K4: FC via WMMA + fused LIFh.  One wave per 16x16 output tile, 493 k-steps
// of v_wmma_f32_16x16x32_f16.  A loaded per ISA A-fragment layout (two 16B
// loads/lane/k-step); B pre-packed (one coalesced 32B load/lane/k-step).
// Prefetch with locality 3 so it fills the WGP cache (SYS-scope prefetch
// would bypass the near cache, which is useless for L2-resident data).
// Epilogue applies bias + LIF on memh and records spikes; each (m,n) is
// owned by exactly one (lane, vgpr) slot.
// ---------------------------------------------------------------------------
__global__ void __launch_bounds__(256)
k_fc_lif(const f16* __restrict__ A, const f16* __restrict__ Bp,
         const float* __restrict__ fc1b, float* __restrict__ memh,
         float* __restrict__ spk_rec_t) {
  int lane = threadIdx.x & 31;
  int wave = threadIdx.x >> 5;
  int tile = blockIdx.x * 8 + wave;   // 256 tiles: 16 m-tiles x 16 n-tiles
  int mt = tile >> 4;
  int nt = tile & 15;

  int m_lane = mt * 16 + (lane & 15);
  const f16* arow = A + (long)m_lane * KDIM + ((lane < 16) ? 0 : 8);
  const f16* bptr = Bp + (long)nt * KS * 512 + lane * 16;

  v8f acc = {0.f, 0.f, 0.f, 0.f, 0.f, 0.f, 0.f, 0.f};
  for (int ks = 0; ks < KS; ++ks) {
    __builtin_prefetch(arow + 4 * 32, 0, 3);   // global_prefetch_b8, WGP-near
    __builtin_prefetch(bptr + 4 * 512, 0, 3);
    v8h alo = *(const v8h*)(arow);             // k-offsets {0..7} / {8..15}
    v8h ahi = *(const v8h*)(arow + 16);        // k-offsets {16..23} / {24..31}
    v16h a = __builtin_shufflevector(alo, ahi, 0, 1, 2, 3, 4, 5, 6, 7,
                                               8, 9, 10, 11, 12, 13, 14, 15);
    v16h b = *(const v16h*)(bptr);
    acc = __builtin_amdgcn_wmma_f32_16x16x32_f16(
        /*neg_a=*/false, a, /*neg_b=*/false, b,
        /*c_mod=*/(short)0, acc, /*reuse_a=*/false, /*reuse_b=*/false);
    arow += 32;
    bptr += 512;
  }

  int n = nt * 16 + (lane & 15);
  int mbase = mt * 16 + ((lane < 16) ? 0 : 8);
  if (n < 250) {
    float bias = fc1b[n];
    #pragma unroll
    for (int r = 0; r < 8; ++r) {
      int m = mbase + r;
      long off = (long)m * 250 + n;
      float mm = memh[off];
      float rst = (mm > THR) ? 1.0f : 0.0f;
      mm = BETA * mm + (acc[r] + bias) - rst * THR;
      memh[off] = mm;
      spk_rec_t[off] = (mm > THR) ? 1.0f : 0.0f;
    }
  }
}

// ---------------------------------------------------------------------------
// K5: spike sums (-> d_out s1/s2) + depthwise temporal conv(7,pad3)+relu +
// adaptive-avg-pool bins [0:13],[12:25],[25:38],[37:50] -> feat [256,1000]
// ---------------------------------------------------------------------------
__global__ void k_head1(const float* __restrict__ spk_rec,
                        const float* __restrict__ tfw, const float* __restrict__ tfb,
                        float* __restrict__ feat, float* __restrict__ out_s) {
  long idx = (long)blockIdx.x * 256 + threadIdx.x;
  const long total = (long)BT * 250;
  if (idx >= total) return;
  int c = (int)(idx % 250);
  int m = (int)(idx / 250);

  float s[NSTEP];
  float ssum = 0.0f;
  for (int t = 0; t < NSTEP; ++t) {
    s[t] = spk_rec[((long)t * BT + m) * 250 + c];
    ssum += s[t];
  }
  float* sp = (m < 128) ? (out_s + (long)m * 250) : (out_s + 32000 + (long)(m - 128) * 250);
  sp[c] = ssum;

  float w[7];
  #pragma unroll
  for (int j = 0; j < 7; ++j) w[j] = tfw[c * 7 + j];
  float bias = tfb[c];

  float y[NSTEP];
  for (int t = 0; t < NSTEP; ++t) {
    float a = bias;
    #pragma unroll
    for (int j = 0; j < 7; ++j) {
      int tt = t + j - 3;
      if (tt >= 0 && tt < NSTEP) a += s[tt] * w[j];
    }
    y[t] = fmaxf(a, 0.0f);
  }
  const int starts[4] = {0, 12, 25, 37};
  #pragma unroll
  for (int b = 0; b < 4; ++b) {
    float a = 0.0f;
    for (int i = 0; i < 13; ++i) a += y[starts[b] + i];
    feat[(long)m * 1000 + c * 4 + b] = a / 13.0f;
  }
}

// K6: h = relu(feat @ fcAw^T + fcAb)   [256,1000]x[1000,256]
__global__ void k_fcA(const float* __restrict__ feat, const float* __restrict__ w,
                      const float* __restrict__ b, float* __restrict__ h) {
  long idx = (long)blockIdx.x * 256 + threadIdx.x;
  if (idx >= (long)BT * 256) return;
  int n = (int)(idx & 255);
  int m = (int)(idx >> 8);
  const float* f = feat + (long)m * 1000;
  const float* wr = w + (long)n * 1000;
  float acc = b[n];
  for (int k = 0; k < 1000; ++k) acc += f[k] * wr[k];
  h[idx] = fmaxf(acc, 0.0f);
}

// K7: o = h @ fcBw^T + fcBb, L2-normalize rows -> emb1/emb2 in d_out
__global__ void __launch_bounds__(128)
k_fcB_norm(const float* __restrict__ h, const float* __restrict__ w,
           const float* __restrict__ b, float* __restrict__ out) {
  __shared__ float red[128];
  int m = blockIdx.x;
  int n = threadIdx.x;
  const float* hr = h + (long)m * 256;
  const float* wr = w + (long)n * 256;
  float acc = b[n];
  for (int k = 0; k < 256; ++k) acc += hr[k] * wr[k];
  red[n] = acc * acc;
  __syncthreads();
  for (int s = 64; s > 0; s >>= 1) {
    if (n < s) red[n] += red[n + s];
    __syncthreads();
  }
  float nrm = fmaxf(sqrtf(red[0]), 1e-12f);
  float* o = (m < 128) ? (out + (long)m * 128) : (out + 16384 + (long)(m - 128) * 128);
  o[n] = acc / nrm;
}

// ---------------------------------------------------------------------------
static inline int cdiv(long a, long b) { return (int)((a + b - 1) / b); }

extern "C" void kernel_launch(void* const* d_in, const int* in_sizes, int n_in,
                              void* d_out, int out_size, void* d_ws, size_t ws_size,
                              hipStream_t stream) {
  (void)in_sizes; (void)n_in; (void)out_size;
  const float* x1   = (const float*)d_in[0];
  const float* x2   = (const float*)d_in[1];
  const float* c1w  = (const float*)d_in[2];
  const float* c1b  = (const float*)d_in[3];
  const float* bn1g = (const float*)d_in[4];
  const float* bn1b = (const float*)d_in[5];
  const float* bn1m = (const float*)d_in[6];
  const float* bn1v = (const float*)d_in[7];
  const float* c2w  = (const float*)d_in[8];
  const float* c2b  = (const float*)d_in[9];
  const float* bn2g = (const float*)d_in[10];
  const float* bn2b = (const float*)d_in[11];
  const float* bn2m = (const float*)d_in[12];
  const float* bn2v = (const float*)d_in[13];
  const float* fc1w = (const float*)d_in[14];
  const float* fc1b = (const float*)d_in[15];
  const float* tfw  = (const float*)d_in[16];
  const float* tfb  = (const float*)d_in[17];
  const float* fcAw = (const float*)d_in[18];
  const float* fcAb = (const float*)d_in[19];
  const float* fcBw = (const float*)d_in[20];
  const float* fcBb = (const float*)d_in[21];
  float* out = (float*)d_out;

  // ---- workspace carve-up (bytes, 256B aligned) ----
  size_t off = 0;
  auto carve = [&](size_t bytes) { size_t o = off; off = (off + bytes + 255) & ~(size_t)255; return o; };
  char* ws = (char*)d_ws;
  const size_t nBP   = 16ul * KS * 512;              // f16
  const size_t nA1   = (size_t)BT * 16 * L1;         // f32
  const size_t nM2   = (size_t)BT * 32 * L2C;        // f32
  const size_t nMH   = (size_t)BT * 250;             // f32
  const size_t nSpk2 = (size_t)BT * KDIM;            // f16
  const size_t nRec  = (size_t)NSTEP * BT * 250;     // f32
  f16*   bp     = (f16*)  (ws + carve(nBP * 2));
  float* a1     = (float*)(ws + carve(nA1 * 4));
  float* mem1   = (float*)(ws + carve(nA1 * 4));
  float* mem2   = (float*)(ws + carve(nM2 * 4));
  float* memh   = (float*)(ws + carve(nMH * 4));
  f16*   spk2   = (f16*)  (ws + carve(nSpk2 * 2));
  float* rec    = (float*)(ws + carve(nRec * 4));
  float* feat   = (float*)(ws + carve((size_t)BT * 1000 * 4));
  float* hbuf   = (float*)(ws + carve((size_t)BT * 256 * 4));
  if (off > ws_size) return;  // ~112 MB required

  // ---- prep ----
  k_pack_b<<<cdiv((long)nBP, 256), 256, 0, stream>>>(fc1w, bp);
  k_conv1<<<cdiv((long)nA1, 256), 256, 0, stream>>>(x1, x2, c1w, c1b, bn1g, bn1b, bn1m, bn1v, a1);
  k_zero<<<2048, 256, 0, stream>>>(mem1, (long)nA1);
  k_zero<<<2048, 256, 0, stream>>>(mem2, (long)nM2);
  k_zero<<<256, 256, 0, stream>>>(memh, (long)nMH);

  // ---- 50-step scan ----
  for (int t = 0; t < NSTEP; ++t) {
    k_step_front<<<BT, 256, 0, stream>>>(a1, mem1, c2w, c2b,
                                         bn2g, bn2b, bn2m, bn2v, mem2, spk2);
    k_fc_lif<<<32, 256, 0, stream>>>(spk2, bp, fc1b, memh, rec + (long)t * BT * 250);
  }

  // ---- head ----
  k_head1<<<cdiv((long)BT * 250, 256), 256, 0, stream>>>(rec, tfw, tfb, feat, out + 32768);
  k_fcA<<<cdiv((long)BT * 256, 256), 256, 0, stream>>>(feat, fcAw, fcAb, hbuf);
  k_fcB_norm<<<BT, 128, 0, stream>>>(hbuf, fcBw, fcBb, out);
}
